// MultiHopAttentionGNN_6313601925386
// MI455X (gfx1250) — matrix operand order repaired and checked by reading, not compiled, not run on hardware
//
#include <hip/hip_runtime.h>
#include <hip/hip_bf16.h>

typedef __attribute__((ext_vector_type(16))) _Float16 v16h;
typedef __attribute__((ext_vector_type(8)))  _Float16 v8h;
typedef __attribute__((ext_vector_type(8)))  float    v8f;

#define NODES 50000       // multiple of 16 -> no GEMM row tail
#define EDGES 300000
#define ETOT  350000      // EDGES + NODES self-loops
#define FDIM  128         // IN = H*HID
#define NHEAD 4
#define CHID  32
#define COUT  256
#define NHOPS 8
#define KCHUNKS 4         // FDIM / 32
#define LRELU_SLOPE 0.2f
#define NEGINF_ORD 0x007FFFFFu   // f2ord(-inf)

// ---------- helpers ----------
__device__ __forceinline__ unsigned f2ord(float f) {
  unsigned u = __float_as_uint(f);
  return (u & 0x80000000u) ? ~u : (u | 0x80000000u);
}
__device__ __forceinline__ float ord2f(unsigned u) {
  unsigned v = (u & 0x80000000u) ? (u & 0x7FFFFFFFu) : ~u;
  return __uint_as_float(v);
}

__global__ void fill_u32_kernel(unsigned* __restrict__ p, unsigned v, size_t n) {
  size_t i = (size_t)blockIdx.x * blockDim.x + threadIdx.x;
  if (i < n) p[i] = v;
}

__global__ void f32_to_f16_kernel(const float* __restrict__ in,
                                  _Float16* __restrict__ out, size_t n) {
  size_t i = (size_t)blockIdx.x * blockDim.x + threadIdx.x;
  if (i < n) out[i] = (_Float16)in[i];
}

// ---------- pack weights into WMMA B-fragment order ----------
// W [K=128, ncols] f32 row-major  ->  Wp f16 laid out as
// [colTile][kChunk][lane][j0..15] so each lane's 16 B-halves are contiguous.
// B 32x16 layout: lane&15 = column; j<8 -> K = 8*(lane>>4)+j, j>=8 -> K = 16+8*(lane>>4)+(j-8).
__global__ void pack_w_kernel(const float* __restrict__ W,
                              _Float16* __restrict__ Wp, int ncols) {
  int p = blockIdx.x * blockDim.x + threadIdx.x;
  if (p >= FDIM * ncols) return;
  int j       = p & 15;
  int lane    = (p >> 4) & 31;
  int chunk   = (p >> 9) & (KCHUNKS - 1);
  int colTile = p >> 11;
  int k   = chunk * 32 + (lane >> 4) * 8 + (j & 7) + (j >> 3) * 16;
  int col = colTile * 16 + (lane & 15);
  Wp[p] = (_Float16)W[(size_t)k * ncols + col];
}

// ---------- WMMA GEMM:  C[nrows,NCOLS] = A_f16[nrows,128] x B(packed)[128,NCOLS] ----------
// One wave -> one 16x16 f32 tile; K = 128 in 4 chunks of 32 via v_wmma_f32_16x16x32_f16.
// nrows must be a multiple of 16 (NODES is). NCOLS compile-time -> constant-offset stores.
template <int NCOLS>
__global__ __launch_bounds__(256)
void gemm_f16_wmma_kernel(const _Float16* __restrict__ A,
                          const _Float16* __restrict__ Wp,
                          float* __restrict__ C, int nrows) {
  const int lane = threadIdx.x & 31;
  const int wave = threadIdx.x >> 5;
  const int rowTile = blockIdx.x * 8 + wave;
  const int row0 = rowTile * 16;
  if (row0 >= nrows) return;          // wave-uniform exit: EXEC all-ones for WMMA

  const int m  = lane & 15;
  const int kq = lane >> 4;
  const _Float16* ap0 = A + (size_t)(row0 + m) * FDIM + kq * 8;
  const _Float16* bp0 = Wp + (((size_t)blockIdx.y * KCHUNKS) * 32 + lane) * 16;

  v8f acc = {};
  #pragma unroll
  for (int c = 0; c < KCHUNKS; ++c) {
    // A fragment (16-bit A 16x32): lane<16 -> K {k0..k0+7, k0+16..k0+23}; lane>=16 -> +8
    union { v16h v; v8h h[2]; } a;
    a.h[0] = *(const v8h*)(ap0 + c * 32);
    a.h[1] = *(const v8h*)(ap0 + c * 32 + 16);
    // B fragment: 16 contiguous halves per lane (pre-packed)
    union { v16h v; v8h h[2]; } b;
    b.h[0] = *(const v8h*)(bp0 + c * 512);
    b.h[1] = *(const v8h*)(bp0 + c * 512 + 8);

    acc = __builtin_amdgcn_wmma_f32_16x16x32_f16(false, a.v, false, b.v,
                                                 (short)0, acc, false, false);
  }
  // C/D layout: VGPR r -> M = r + 8*kq, N = lane&15; constant-offset b32 stores
  float* cp = C + (size_t)(row0 + 8 * kq) * NCOLS + blockIdx.y * 16 + m;
  #pragma unroll
  for (int r = 0; r < 8; ++r) cp[r * NCOLS] = acc[r];
}

// ---------- per-node attention dots: e_src/e_dst [N,H] ----------
__global__ void attn_scores_kernel(const float* __restrict__ h,
                                   const float* __restrict__ asrc,
                                   const float* __restrict__ adst,
                                   float* __restrict__ esrc,
                                   float* __restrict__ edst, int cc) {
  int t = blockIdx.x * blockDim.x + threadIdx.x;
  if (t >= NODES * NHEAD) return;
  int node = t / NHEAD, hd = t % NHEAD;
  const float* hr = h + (size_t)node * NHEAD * cc + (size_t)hd * cc;
  const float* as = asrc + (size_t)hd * cc;
  const float* ad = adst + (size_t)hd * cc;
  float ss = 0.f, sd = 0.f;
  for (int c = 0; c < cc; ++c) { float v = hr[c]; ss += v * as[c]; sd += v * ad[c]; }
  esrc[t] = ss;
  edst[t] = sd;
}

// ---------- edge pass 1: leaky-relu score + segment max (ordered-uint atomicMax) ----------
__global__ void edge_score_max_kernel(const float* __restrict__ esrc,
                                      const float* __restrict__ edst,
                                      const int* __restrict__ src,
                                      const int* __restrict__ dst,
                                      float* __restrict__ ebuf,
                                      unsigned* __restrict__ mbuf) {
  int t = blockIdx.x * blockDim.x + threadIdx.x;
  if (t >= ETOT * NHEAD) return;
  int e = t >> 2, hd = t & 3;
  int s = (e < EDGES) ? src[e] : (e - EDGES);
  int d = (e < EDGES) ? dst[e] : (e - EDGES);
  float v = esrc[s * NHEAD + hd] + edst[d * NHEAD + hd];
  v = (v > 0.f) ? v : LRELU_SLOPE * v;
  ebuf[t] = v;
  atomicMax(&mbuf[d * NHEAD + hd], f2ord(v));
}

// ---------- edge pass 2: p = exp(e - m), segment sum ----------
__global__ void edge_exp_sum_kernel(const int* __restrict__ dst,
                                    float* __restrict__ ebuf,
                                    const unsigned* __restrict__ mbuf,
                                    float* __restrict__ denom) {
  int t = blockIdx.x * blockDim.x + threadIdx.x;
  if (t >= ETOT * NHEAD) return;
  int e = t >> 2, hd = t & 3;
  int d = (e < EDGES) ? dst[e] : (e - EDGES);
  float m = ord2f(mbuf[d * NHEAD + hd]);
  float p = __expf(ebuf[t] - m);
  ebuf[t] = p;
  atomicAdd(&denom[d * NHEAD + hd], p);
}

// ---------- edge pass 3: agg[dst] += (p/denom[dst]) * h[src] ----------
__global__ void edge_aggregate_kernel(const int* __restrict__ src,
                                      const int* __restrict__ dst,
                                      const float* __restrict__ ebuf,
                                      const float* __restrict__ denom,
                                      const float* __restrict__ h,
                                      float* __restrict__ agg, int cc) {
  int t = blockIdx.x * blockDim.x + threadIdx.x;
  if (t >= ETOT * NHEAD) return;
  int e = t >> 2, hd = t & 3;
  int s = (e < EDGES) ? src[e] : (e - EDGES);
  int d = (e < EDGES) ? dst[e] : (e - EDGES);
  float alpha = ebuf[t] / denom[d * NHEAD + hd];
  const float* hr = h   + (size_t)s * NHEAD * cc + (size_t)hd * cc;
  float*       ar = agg + (size_t)d * NHEAD * cc + (size_t)hd * cc;
  for (int c = 0; c < cc; ++c) atomicAdd(&ar[c], alpha * hr[c]);
}

// ---------- hop epilogue: relu(agg + bias) -> f16 features for next hop ----------
__global__ void finalize_hop_kernel(const float* __restrict__ agg,
                                    const float* __restrict__ bias,
                                    _Float16* __restrict__ xh) {
  int t = blockIdx.x * blockDim.x + threadIdx.x;
  if (t >= NODES * FDIM) return;
  float v = agg[t] + bias[t & (FDIM - 1)];
  v = (v > 0.f) ? v : 0.f;
  xh[t] = (_Float16)v;
}

// ---------- output epilogue: mean over heads + bias ----------
__global__ void finalize_out_kernel(const float* __restrict__ agg,
                                    const float* __restrict__ bias,
                                    float* __restrict__ out) {
  int t = blockIdx.x * blockDim.x + threadIdx.x;
  if (t >= NODES * COUT) return;
  int n = t / COUT, c = t % COUT;
  const float* a = agg + (size_t)n * NHEAD * COUT;
  float s = 0.f;
  #pragma unroll
  for (int hd = 0; hd < NHEAD; ++hd) s += a[hd * COUT + c];
  out[t] = 0.25f * s + bias[c];
}

// ---------- host orchestration ----------
static inline int cdiv(long long a, long long b) { return (int)((a + b - 1) / b); }

extern "C" void kernel_launch(void* const* d_in, const int* in_sizes, int n_in,
                              void* d_out, int out_size, void* d_ws, size_t ws_size,
                              hipStream_t stream) {
  const float* x_in        = (const float*)d_in[0];
  const int*   ei          = (const int*)d_in[1];     // [2,E]
  const float* Ws          = (const float*)d_in[2];   // [HOPS,128,128]
  const float* att_src     = (const float*)d_in[3];   // [HOPS,4,32]
  const float* att_dst     = (const float*)d_in[4];
  const float* biases      = (const float*)d_in[5];   // [HOPS,128]
  const float* W_out       = (const float*)d_in[6];   // [128,1024]
  const float* att_src_out = (const float*)d_in[7];   // [4,256]
  const float* att_dst_out = (const float*)d_in[8];
  const float* bias_out    = (const float*)d_in[9];   // [256]
  float*       out         = (float*)d_out;
  const int* src = ei;
  const int* dst = ei + EDGES;

  // workspace layout
  char* ws = (char*)d_ws;
  size_t off = 0;
  auto take = [&](size_t bytes) -> char* {
    char* p = ws + off;
    off += (bytes + 255) & ~(size_t)255;
    return p;
  };
  _Float16* xh    = (_Float16*)take((size_t)NODES * FDIM * sizeof(_Float16));
  _Float16* Wp    = (_Float16*)take((size_t)FDIM * NHEAD * COUT * sizeof(_Float16));
  float*    hbuf  = (float*)take((size_t)NODES * NHEAD * COUT * sizeof(float));
  float*    agg   = (float*)take((size_t)NODES * NHEAD * COUT * sizeof(float));
  float*    ebuf  = (float*)take((size_t)ETOT * NHEAD * sizeof(float));
  unsigned* mbuf  = (unsigned*)take((size_t)NODES * NHEAD * sizeof(unsigned));
  float*    denom = (float*)take((size_t)NODES * NHEAD * sizeof(float));
  float*    esrc  = (float*)take((size_t)NODES * NHEAD * sizeof(float));
  float*    edstb = (float*)take((size_t)NODES * NHEAD * sizeof(float));

  const int TB = 256;
  const int rowBlocks = cdiv(cdiv(NODES, 16), 8);     // 8 waves/block, 16 rows/wave
  const int edgeGrid  = cdiv((long long)ETOT * NHEAD, TB);
  const int nodeHGrid = cdiv((long long)NODES * NHEAD, TB);

  // initial feature conversion to f16
  f32_to_f16_kernel<<<cdiv((long long)NODES * FDIM, TB), TB, 0, stream>>>(
      x_in, xh, (size_t)NODES * FDIM);

  auto run_layer = [&](const float* W, const float* as, const float* ad,
                       int ncols, int cc) {
    // weights -> f16, pre-packed in B-fragment order
    pack_w_kernel<<<cdiv((long long)FDIM * ncols, TB), TB, 0, stream>>>(W, Wp, ncols);
    // h = xh @ W  (WMMA)
    dim3 grid(rowBlocks, ncols / 16);
    if (ncols == FDIM)
      gemm_f16_wmma_kernel<FDIM><<<grid, TB, 0, stream>>>(xh, Wp, hbuf, NODES);
    else
      gemm_f16_wmma_kernel<NHEAD * COUT><<<grid, TB, 0, stream>>>(xh, Wp, hbuf, NODES);
    // per-node attention dots
    attn_scores_kernel<<<nodeHGrid, TB, 0, stream>>>(hbuf, as, ad, esrc, edstb, cc);
    // init segment buffers
    fill_u32_kernel<<<cdiv((long long)NODES * NHEAD, TB), TB, 0, stream>>>(
        mbuf, NEGINF_ORD, (size_t)NODES * NHEAD);
    fill_u32_kernel<<<cdiv((long long)NODES * NHEAD, TB), TB, 0, stream>>>(
        (unsigned*)denom, 0u, (size_t)NODES * NHEAD);
    fill_u32_kernel<<<cdiv((long long)NODES * NHEAD * cc, TB), TB, 0, stream>>>(
        (unsigned*)agg, 0u, (size_t)NODES * NHEAD * cc);
    // edge phase
    edge_score_max_kernel<<<edgeGrid, TB, 0, stream>>>(esrc, edstb, src, dst, ebuf, mbuf);
    edge_exp_sum_kernel<<<edgeGrid, TB, 0, stream>>>(dst, ebuf, mbuf, denom);
    edge_aggregate_kernel<<<edgeGrid, TB, 0, stream>>>(src, dst, ebuf, denom, hbuf, agg, cc);
  };

  // 8 hop layers (concat=True, relu)
  for (int hop = 0; hop < NHOPS; ++hop) {
    run_layer(Ws + (size_t)hop * FDIM * FDIM,
              att_src + (size_t)hop * NHEAD * CHID,
              att_dst + (size_t)hop * NHEAD * CHID,
              FDIM, CHID);
    finalize_hop_kernel<<<cdiv((long long)NODES * FDIM, TB), TB, 0, stream>>>(
        agg, biases + (size_t)hop * FDIM, xh);
  }

  // output layer (concat=False -> mean over heads)
  run_layer(W_out, att_src_out, att_dst_out, NHEAD * COUT, COUT);
  finalize_out_kernel<<<cdiv((long long)NODES * COUT, TB), TB, 0, stream>>>(
      agg, bias_out, out);
}
